// EventMotionModel_38044820308140
// MI455X (gfx1250) — matrix operand order repaired and checked by reading, not compiled, not run on hardware
//
#include <hip/hip_runtime.h>
#include <cstdint>
#include <cstddef>

// ---------------------------------------------------------------------------
// EventMotionModel forward for MI455X (gfx1250, wave32, WMMA bf16)
// ---------------------------------------------------------------------------

typedef __attribute__((ext_vector_type(16))) __bf16 v16bf;
typedef __attribute__((ext_vector_type(8)))  __bf16 v8bf;
typedef __attribute__((ext_vector_type(8)))  float  v8f;

#define N_TOK   65536        // B*T = 2048*32
#define H_DIM   1024
#define D_IN    512
#define C_DIM   256
#define D_CAT   768          // IN + CD
#define F_HID   128
#define O_DIM   512
#define T_SEQ   32
#define D_HEAD  128

__device__ __forceinline__ float bf2f(__bf16 b) {
  union { unsigned short s; __bf16 b; } i; i.b = b;
  union { uint32_t u; float f; } o; o.u = ((uint32_t)i.s) << 16;
  return o.f;
}
__device__ __forceinline__ __bf16 f2bf(float f) {
  union { float f; uint32_t u; } i; i.f = f;
  uint32_t r = (i.u + 0x7FFFu + ((i.u >> 16) & 1u)) >> 16;   // RNE
  union { unsigned short s; __bf16 b; } o; o.s = (unsigned short)r;
  return o.b;
}

// --------------------------- weight f32 -> bf16^T ---------------------------
// src: [K, Nc] row-major f32 ; dst: [Nc, K] row-major bf16 (out-major)
__global__ __launch_bounds__(256)
void convw_k(const float* __restrict__ w, __bf16* __restrict__ wt, int K, int Nc) {
  size_t gid = (size_t)blockIdx.x * 256 + threadIdx.x;
  if (gid >= (size_t)K * Nc) return;
  int o = (int)(gid / K);
  int k = (int)(gid % K);
  wt[gid] = f2bf(w[(size_t)k * Nc + o]);
}

// --------------------- input concat + clip + bf16 cast ----------------------
__global__ __launch_bounds__(256)
void prep_k(const float* __restrict__ state, const float* __restrict__ cond,
            __bf16* __restrict__ xb, __bf16* __restrict__ condb) {
  size_t gid = (size_t)blockIdx.x * 256 + threadIdx.x;
  if (gid >= (size_t)N_TOK * D_CAT) return;
  size_t row = gid / D_CAT;
  int c = (int)(gid % D_CAT);
  float v = (c < D_IN) ? state[row * D_IN + c] : cond[row * C_DIM + (c - D_IN)];
  float cl = fminf(fmaxf(v, -16.0f), 16.0f);
  xb[row * H_DIM + c] = f2bf(cl);
  if (c >= D_IN) condb[row * C_DIM + (c - D_IN)] = f2bf(v); // FiLM uses raw cond
}

// ------------------------------ fused GEMM ----------------------------------
// C[16 x NT*128] per block. 8 waves; wave owns NT 16-col WMMA tiles.
// A: [rows, lda] bf16 row-major.  Wt: [out, K] bf16 (transposed weights).
// EPI: 0 none->bf16 | 1 bias | 2 bias+relu | 3 bias+gelu | 4 bias+(1+.5tanh)
//      5 bias+LN+relu | 6 +add1+add2 then LN | 7 bias->f32 | 8 bias+(.5tanh)
template<int NT, int EPI>
__global__ __launch_bounds__(256)
void gemm_k(const __bf16* __restrict__ A, int lda, int K,
            const __bf16* __restrict__ Wt,
            const float* __restrict__ bias,
            const float* __restrict__ gamma,
            const float* __restrict__ beta,
            const __bf16* __restrict__ add1,
            const __bf16* __restrict__ add2,
            __bf16* __restrict__ outb,
            float*  __restrict__ outf,
            int out_stride, int col_off) {
  constexpr int  CPB      = NT * 128;
  constexpr bool LN       = (EPI == 5 || EPI == 6);
  constexpr bool HAS_BIAS = (EPI != 0 && EPI != 6);

  const int rb0  = blockIdx.x * 16;
  const int wave = threadIdx.x >> 5;
  const int lane = threadIdx.x & 31;
  const int l15  = lane & 15;
  const int half = (lane >> 4) & 1;   // 0: K {0..7,16..23}/{0..15}, 1: high halves

  v8f acc[NT];
#pragma unroll
  for (int t = 0; t < NT; ++t)
#pragma unroll
    for (int j = 0; j < 8; ++j) acc[t][j] = 0.0f;

  // A operand: row = l15, per-lane 16 contiguous-ish bf16 (two 16B loads)
  const __bf16* Arow = A + (size_t)(rb0 + l15) * lda + (half ? 16 : 0);
  const int wbase = col_off + wave * (NT * 16) + l15;

  for (int kt = 0; kt < K; kt += 32) {
    v8bf alo = *(const v8bf*)(Arow + kt);
    v8bf ahi = *(const v8bf*)(Arow + kt + 8);
    v16bf a = __builtin_shufflevector(alo, ahi, 0,1,2,3,4,5,6,7,8,9,10,11,12,13,14,15);
#pragma unroll
    for (int t = 0; t < NT; ++t) {
      // B operand: lane = column n, 16 contiguous K values from transposed W
      const v16bf* bp = (const v16bf*)(Wt + (size_t)(wbase + t * 16) * K
                                          + kt + (half ? 16 : 0));
      acc[t] = __builtin_amdgcn_wmma_f32_16x16x32_bf16(
                 false, a, false, *bp, (short)0, acc[t], false, false);
    }
  }

  __shared__ float red[32];   // [0..15] sum, [16..31] sumsq per row
  __shared__ float mvar[32];  // [0..15] mean, [16..31] rstd

  if constexpr (LN) {
    if (threadIdx.x < 32) red[threadIdx.x] = 0.0f;
    __syncthreads();
#pragma unroll
    for (int j = 0; j < 8; ++j) {
      const int rloc = j + half * 8;
      const size_t grow = (size_t)(rb0 + rloc);
      float s = 0.f, q = 0.f;
#pragma unroll
      for (int t = 0; t < NT; ++t) {
        const int lc = wave * (NT * 16) + t * 16 + l15;
        float v = acc[t][j];
        if constexpr (HAS_BIAS) v += bias[lc + col_off];
        if constexpr (EPI == 6)
          v += bf2f(add1[grow * H_DIM + lc]) + bf2f(add2[grow * H_DIM + lc]);
        acc[t][j] = v;
        s += v; q += v * v;
      }
      atomicAdd(&red[rloc], s);        // ds_add_f32
      atomicAdd(&red[16 + rloc], q);
    }
    __syncthreads();
    if (threadIdx.x < 16) {
      float m   = red[threadIdx.x] * (1.0f / CPB);
      float var = red[16 + threadIdx.x] * (1.0f / CPB) - m * m;
      mvar[threadIdx.x]      = m;
      mvar[16 + threadIdx.x] = rsqrtf(var + 1e-5f);
    }
    __syncthreads();
  }

#pragma unroll
  for (int j = 0; j < 8; ++j) {
    const int rloc = j + half * 8;
    const size_t grow = (size_t)(rb0 + rloc);
#pragma unroll
    for (int t = 0; t < NT; ++t) {
      const int lc = wave * (NT * 16) + t * 16 + l15;
      float v = acc[t][j];
      if constexpr (!LN && HAS_BIAS) v += bias[lc + col_off];
      if constexpr (EPI == 2) v = fmaxf(v, 0.0f);
      if constexpr (EPI == 3) v = 0.5f * v * (1.0f + erff(v * 0.70710678118f));
      if constexpr (EPI == 4) v = 1.0f + 0.5f * tanhf(v);
      if constexpr (EPI == 8) v = 0.5f * tanhf(v);
      if constexpr (LN) {
        v = (v - mvar[rloc]) * mvar[16 + rloc] * gamma[lc] + beta[lc];
        if constexpr (EPI == 5) v = fmaxf(v, 0.0f);
      }
      if constexpr (EPI == 7) outf[grow * (size_t)out_stride + lc] = v;
      else                    outb[grow * (size_t)out_stride + lc] = f2bf(v);
    }
  }
}

// ------------------- FiLM-gated query LN (per-token) ------------------------
__global__ __launch_bounds__(256)
void qin_k(const __bf16* __restrict__ h, const __bf16* __restrict__ gg,
           const __bf16* __restrict__ bb, const float* __restrict__ lng,
           const float* __restrict__ lnb, __bf16* __restrict__ qout) {
  const size_t row = blockIdx.x;
  const int tid = threadIdx.x;
  __shared__ float rs[256], rq[256];
  float v[4]; float s = 0.f, q = 0.f;
#pragma unroll
  for (int i = 0; i < 4; ++i) {
    int c = tid + 256 * i;
    v[i] = bf2f(h[row * H_DIM + c]);
    s += v[i]; q += v[i] * v[i];
  }
  rs[tid] = s; rq[tid] = q;
  __syncthreads();
  for (int o = 128; o > 0; o >>= 1) {
    if (tid < o) { rs[tid] += rs[tid + o]; rq[tid] += rq[tid + o]; }
    __syncthreads();
  }
  const float mean = rs[0] * (1.0f / H_DIM);
  const float rstd = rsqrtf(rq[0] * (1.0f / H_DIM) - mean * mean + 1e-5f);
#pragma unroll
  for (int i = 0; i < 4; ++i) {
    int c = tid + 256 * i;
    float x = (v[i] - mean) * rstd * lng[c] + lnb[c];
    float g = bf2f(gg[row * H_DIM + c]);   // already 1 + 0.5*tanh
    float b = bf2f(bb[row * H_DIM + c]);   // already 0.5*tanh
    qout[row * H_DIM + c] = f2bf(x * g + b);
  }
}

// --------------------- attention: one (batch, head) per block ---------------
__global__ __launch_bounds__(256)
void attn_k(const __bf16* __restrict__ qb, const __bf16* __restrict__ kb,
            const __bf16* __restrict__ vb, __bf16* __restrict__ ob) {
  __shared__ float qs[T_SEQ * D_HEAD];
  __shared__ float ks[T_SEQ * D_HEAD];
  __shared__ float vs[T_SEQ * D_HEAD];
  __shared__ float sc[T_SEQ * T_SEQ];
  const int b   = blockIdx.x >> 3;
  const int hd  = blockIdx.x & 7;
  const int tid = threadIdx.x;
  const size_t base = ((size_t)b * T_SEQ) * H_DIM + (size_t)hd * D_HEAD;
  for (int e = tid; e < T_SEQ * D_HEAD; e += 256) {
    int t = e >> 7, d = e & 127;
    size_t g = base + (size_t)t * H_DIM + d;
    qs[e] = bf2f(qb[g]); ks[e] = bf2f(kb[g]); vs[e] = bf2f(vb[g]);
  }
  __syncthreads();
  for (int e = tid; e < T_SEQ * T_SEQ; e += 256) {
    int i = e >> 5, j = e & 31;
    float s = 0.f;
    for (int d = 0; d < D_HEAD; ++d) s += qs[i * D_HEAD + d] * ks[j * D_HEAD + d];
    sc[e] = s * 0.08838834764831845f;   // 1/sqrt(128)
  }
  __syncthreads();
  if (tid < T_SEQ) {
    float mx = -1e30f;
    for (int j = 0; j < T_SEQ; ++j) mx = fmaxf(mx, sc[tid * T_SEQ + j]);
    float sum = 0.f;
    for (int j = 0; j < T_SEQ; ++j) {
      float e2 = expf(sc[tid * T_SEQ + j] - mx);
      sc[tid * T_SEQ + j] = e2; sum += e2;
    }
    float inv = 1.0f / sum;
    for (int j = 0; j < T_SEQ; ++j) sc[tid * T_SEQ + j] *= inv;
  }
  __syncthreads();
  for (int e = tid; e < T_SEQ * D_HEAD; e += 256) {
    int i = e >> 7, d = e & 127;
    float s = 0.f;
    for (int j = 0; j < T_SEQ; ++j) s += sc[i * T_SEQ + j] * vs[j * D_HEAD + d];
    ob[base + (size_t)i * H_DIM + d] = f2bf(s);
  }
}

// ---------------------------------------------------------------------------
extern "C" void kernel_launch(void* const* d_in, const int* in_sizes, int n_in,
                              void* d_out, int out_size, void* d_ws, size_t ws_size,
                              hipStream_t stream) {
  (void)in_sizes; (void)n_in; (void)out_size; (void)ws_size;
  const float* state = (const float*)d_in[0];
  const float* cond  = (const float*)d_in[1];
  const float* ew1 = (const float*)d_in[2];   const float* eb1 = (const float*)d_in[3];
  const float* eg1 = (const float*)d_in[4];   const float* ebt1= (const float*)d_in[5];
  const float* ew2 = (const float*)d_in[6];   const float* eb2 = (const float*)d_in[7];
  const float* eg2 = (const float*)d_in[8];   const float* ebt2= (const float*)d_in[9];
  const float* rw  = (const float*)d_in[10];  const float* rb  = (const float*)d_in[11];
  const float* fw1 = (const float*)d_in[12];  const float* fb1 = (const float*)d_in[13];
  const float* fw2 = (const float*)d_in[14];  const float* fb2 = (const float*)d_in[15];
  const float* lnqg= (const float*)d_in[16];  const float* lnqb= (const float*)d_in[17];
  const float* wq  = (const float*)d_in[18];  const float* wk  = (const float*)d_in[19];
  const float* wv  = (const float*)d_in[20];  const float* wo  = (const float*)d_in[21];
  const float* cng = (const float*)d_in[22];  const float* cnb = (const float*)d_in[23];
  const float* hw1 = (const float*)d_in[24];  const float* hb1 = (const float*)d_in[25];
  const float* hw2 = (const float*)d_in[26];  const float* hb2 = (const float*)d_in[27];
  float* out = (float*)d_out;

  char* ws = (char*)d_ws;
  size_t off = 0;
  auto take = [&](size_t elems) -> __bf16* {
    __bf16* p = (__bf16*)(ws + off);
    off += (elems * sizeof(__bf16) + 255) & ~(size_t)255;
    return p;
  };
  // transposed bf16 weights
  __bf16* wt_ew1 = take((size_t)768 * 1024);
  __bf16* wt_ew2 = take((size_t)1024 * 1024);
  __bf16* wt_rw  = take((size_t)768 * 1024);
  __bf16* wt_fw1 = take((size_t)256 * 128);
  __bf16* wt_fw2 = take((size_t)128 * 2048);
  __bf16* wt_wq  = take((size_t)1024 * 1024);
  __bf16* wt_wk  = take((size_t)1024 * 1024);
  __bf16* wt_wv  = take((size_t)1024 * 1024);
  __bf16* wt_wo  = take((size_t)1024 * 1024);
  __bf16* wt_hw1 = take((size_t)1024 * 1024);
  __bf16* wt_hw2 = take((size_t)1024 * 512);
  // token-sized slots (N x 1024 bf16), heavily reused
  const size_t TS = (size_t)N_TOK * H_DIM;
  __bf16* S0 = take(TS);  // xb -> FiLM b-half -> attn output
  __bf16* S1 = take(TS);  // h1 -> FiLM g-half -> q
  __bf16* S2 = take(TS);  // h
  __bf16* S3 = take(TS);  // res
  __bf16* S4 = take(TS);  // qin -> k -> h2
  __bf16* S5 = take(TS);  // v  -> m
  __bf16* condb = take((size_t)N_TOK * C_DIM);
  __bf16* fh    = take((size_t)N_TOK * F_HID);

  auto blks = [](size_t n) { return (unsigned)((n + 255) / 256); };

  // 0) weight conversion (f32 [K,Nc] -> bf16 [Nc,K])
  convw_k<<<blks((size_t)768*1024), 256, 0, stream>>>(ew1, wt_ew1, 768, 1024);
  convw_k<<<blks((size_t)1024*1024),256, 0, stream>>>(ew2, wt_ew2, 1024, 1024);
  convw_k<<<blks((size_t)768*1024), 256, 0, stream>>>(rw,  wt_rw,  768, 1024);
  convw_k<<<blks((size_t)256*128),  256, 0, stream>>>(fw1, wt_fw1, 256, 128);
  convw_k<<<blks((size_t)128*2048), 256, 0, stream>>>(fw2, wt_fw2, 128, 2048);
  convw_k<<<blks((size_t)1024*1024),256, 0, stream>>>(wq,  wt_wq,  1024, 1024);
  convw_k<<<blks((size_t)1024*1024),256, 0, stream>>>(wk,  wt_wk,  1024, 1024);
  convw_k<<<blks((size_t)1024*1024),256, 0, stream>>>(wv,  wt_wv,  1024, 1024);
  convw_k<<<blks((size_t)1024*1024),256, 0, stream>>>(wo,  wt_wo,  1024, 1024);
  convw_k<<<blks((size_t)1024*1024),256, 0, stream>>>(hw1, wt_hw1, 1024, 1024);
  convw_k<<<blks((size_t)1024*512), 256, 0, stream>>>(hw2, wt_hw2, 1024, 512);

  // 1) clip + concat + bf16 inputs
  prep_k<<<blks((size_t)N_TOK * D_CAT), 256, 0, stream>>>(state, cond, S0, condb);

  const unsigned RG = N_TOK / 16;  // 4096 row-blocks

  // 2) encoder1: h1 = relu(LN(xb @ ew1 + eb1))
  gemm_k<8,5><<<RG,256,0,stream>>>(S0, H_DIM, 768,  wt_ew1, eb1, eg1, ebt1,
                                   nullptr, nullptr, S1, nullptr, H_DIM, 0);
  // 3) encoder2: h = relu(LN(h1 @ ew2 + eb2))
  gemm_k<8,5><<<RG,256,0,stream>>>(S1, H_DIM, 1024, wt_ew2, eb2, eg2, ebt2,
                                   nullptr, nullptr, S2, nullptr, H_DIM, 0);
  // 4) residual: res = xb @ rw + rb
  gemm_k<8,1><<<RG,256,0,stream>>>(S0, H_DIM, 768,  wt_rw,  rb, nullptr, nullptr,
                                   nullptr, nullptr, S3, nullptr, H_DIM, 0);
  // 5) FiLM hidden: fh = gelu(cond @ fw1 + fb1)
  gemm_k<1,3><<<RG,256,0,stream>>>(condb, C_DIM, 256, wt_fw1, fb1, nullptr, nullptr,
                                   nullptr, nullptr, fh, nullptr, F_HID, 0);
  // 6) FiLM g (cols 0..1023): 1 + 0.5*tanh
  gemm_k<8,4><<<RG,256,0,stream>>>(fh, F_HID, 128, wt_fw2, fb2, nullptr, nullptr,
                                   nullptr, nullptr, S1, nullptr, H_DIM, 0);
  // 7) FiLM b (cols 1024..2047): 0.5*tanh
  gemm_k<8,8><<<RG,256,0,stream>>>(fh, F_HID, 128, wt_fw2, fb2, nullptr, nullptr,
                                   nullptr, nullptr, S0, nullptr, H_DIM, 1024);
  // 8) qin = LN(h)*g + b
  qin_k<<<N_TOK,256,0,stream>>>(S2, S1, S0, lnqg, lnqb, S4);
  // 9) q / k / v projections (no bias)
  gemm_k<8,0><<<RG,256,0,stream>>>(S4, H_DIM, 1024, wt_wq, nullptr, nullptr, nullptr,
                                   nullptr, nullptr, S1, nullptr, H_DIM, 0);
  gemm_k<8,0><<<RG,256,0,stream>>>(S2, H_DIM, 1024, wt_wk, nullptr, nullptr, nullptr,
                                   nullptr, nullptr, S4, nullptr, H_DIM, 0);
  gemm_k<8,0><<<RG,256,0,stream>>>(S2, H_DIM, 1024, wt_wv, nullptr, nullptr, nullptr,
                                   nullptr, nullptr, S5, nullptr, H_DIM, 0);
  // 10) attention (2048 batches x 8 heads)
  attn_k<<<2048 * 8, 256, 0, stream>>>(S1, S4, S5, S0);
  // 11) h2 = LN(h + attn@wo + res)
  gemm_k<8,6><<<RG,256,0,stream>>>(S0, H_DIM, 1024, wt_wo, nullptr, cng, cnb,
                                   S2, S3, S4, nullptr, H_DIM, 0);
  // 12) m = relu(h2 @ hw1 + hb1)
  gemm_k<8,2><<<RG,256,0,stream>>>(S4, H_DIM, 1024, wt_hw1, hb1, nullptr, nullptr,
                                   nullptr, nullptr, S5, nullptr, H_DIM, 0);
  // 13) out = m @ hw2 + hb2  (f32)
  gemm_k<4,7><<<RG,256,0,stream>>>(S5, H_DIM, 1024, wt_hw2, hb2, nullptr, nullptr,
                                   nullptr, nullptr, nullptr, out, O_DIM, 0);
}
